// Attention_42047729828493
// MI455X (gfx1250) — matrix operand order repaired and checked by reading, not compiled, not run on hardware
//
#include <hip/hip_runtime.h>
#include <hip/hip_bf16.h>

// Problem constants (match reference)
#define B_   2
#define L_   2048
#define H_   1024
#define NH_  16
#define D_   64
#define BL_  (B_*L_)

typedef __attribute__((ext_vector_type(16))) __bf16 v16bf;
typedef __attribute__((ext_vector_type(8)))  float  v8f;

union Frag { unsigned int u[8]; v16bf v; };

__device__ __forceinline__ unsigned short f2bf(float f) {
    unsigned int u = __float_as_uint(f);
    u += 0x7FFFu + ((u >> 16) & 1u);          // round-to-nearest-even
    return (unsigned short)(u >> 16);
}

// ---------------------------------------------------------------------------
// gfx1250 async copy: global -> LDS, 16 bytes, tracked by ASYNCcnt.
// LDS operand = low 32 bits of the generic pointer (ISA 10.2: LDS aperture
// maps addr[31:0] to the LDS offset).
// ---------------------------------------------------------------------------
__device__ __forceinline__ void async_copy_b128(void* ldst, const void* gsrc) {
    unsigned int l = (unsigned int)(unsigned long long)ldst;
    asm volatile("global_load_async_to_lds_b128 %0, %1, off"
                 :: "v"(l), "v"((unsigned long long)gsrc)
                 : "memory");
}
__device__ __forceinline__ void wait_async_all() {
    asm volatile("s_wait_asynccnt 0" ::: "memory");
}

// ---------------------------------------------------------------------------
// Stage 0a: f32 -> bf16 conversion (vectorized: 8 elems / thread / iter)
// ---------------------------------------------------------------------------
__global__ void __launch_bounds__(256)
f32_to_bf16_vec(const float* __restrict__ in, unsigned short* __restrict__ out, int n8) {
    int i = blockIdx.x * blockDim.x + threadIdx.x;
    int stride = gridDim.x * blockDim.x;
    for (; i < n8; i += stride) {
        float4 a = ((const float4*)in)[2 * i];
        float4 b = ((const float4*)in)[2 * i + 1];
        __align__(16) unsigned short s[8] = {
            f2bf(a.x), f2bf(a.y), f2bf(a.z), f2bf(a.w),
            f2bf(b.x), f2bf(b.y), f2bf(b.z), f2bf(b.w)};
        ((uint4*)out)[i] = *(const uint4*)s;
    }
}

// ---------------------------------------------------------------------------
// Stage 0b: weight convert + transpose:  in[K][N] f32  ->  out[N][K] bf16
// (one-time cost; lets the GEMM stage B tiles with pure 16B copies)
// ---------------------------------------------------------------------------
__global__ void __launch_bounds__(256)
f32_to_bf16_transpose(const float* __restrict__ in, unsigned short* __restrict__ out,
                      int Kd, int Nd) {
    int n = blockIdx.x * 32 + (threadIdx.x & 31);
    int k = blockIdx.y * 32 + (threadIdx.x >> 5) * 4;
    #pragma unroll
    for (int i = 0; i < 4; ++i)
        out[(size_t)n * Kd + k + i] = f2bf(in[(size_t)(k + i) * Nd + n]);
}

// ---------------------------------------------------------------------------
// Stage 1/3: bf16 GEMM  C[M,N] = A[M,K] * W[K,N], W given transposed [N][K].
//   mode 0: f32 [M][N]   (final projection into d_out)
//   mode 1: bf16 [M][N]  (Q / K / attn intermediate)
//   mode 2: bf16 per-head transposed [b][h][d][l]  (V, for the PV operand)
// Block tile 128x128, 8 waves, each wave 32x64 (2x4 WMMA accumulators).
// ---------------------------------------------------------------------------
#define LDA 40   // padded LDS stride (ushorts) for 32-wide K tiles

__global__ void __launch_bounds__(256)
gemm_bf16_wmma(const unsigned short* __restrict__ A,
               const unsigned short* __restrict__ Bt,
               void* __restrict__ Cout,
               int M, int N, int K, int mode)
{
    __shared__ __align__(16) unsigned short As[128 * LDA];   // [m][k]
    __shared__ __align__(16) unsigned short Bs[128 * LDA];   // [n][k]

    const int t    = threadIdx.x;
    const int lane = t & 31;
    const int wave = t >> 5;
    const int kh   = lane >> 4;   // half-wave (0/1)
    const int ln   = lane & 15;
    const int wm   = wave & 3;    // wave m-tile (0..3) -> rows wm*32
    const int wn   = wave >> 2;   // wave n-tile (0..1) -> cols wn*64
    const int mblock = blockIdx.y * 128;
    const int nblock = blockIdx.x * 128;

    v8f acc[2][4] = {};

    // cooperative-load coordinates: each thread copies 32B of A and 32B of B
    const int crow = t >> 1;            // 0..127
    const int ccol = (t & 1) * 16;      // 0 or 16

    const unsigned short* aptr = &A [(size_t)(mblock + crow) * K + ccol];
    const unsigned short* bptr = &Bt[(size_t)(nblock + crow) * K + ccol];
    unsigned short* asl = &As[crow * LDA + ccol];
    unsigned short* bsl = &Bs[crow * LDA + ccol];

    for (int kt = 0; kt < K; kt += 32) {
        async_copy_b128(asl,     aptr + kt);
        async_copy_b128(asl + 8, aptr + kt + 8);
        async_copy_b128(bsl,     bptr + kt);
        async_copy_b128(bsl + 8, bptr + kt + 8);
        if (kt + 32 < K) {
            __builtin_prefetch(aptr + kt + 32, 0, 1);   // global_prefetch_b8
            __builtin_prefetch(bptr + kt + 32, 0, 1);
        }
        wait_async_all();
        __syncthreads();

        Frag a[2], b[4];
        #pragma unroll
        for (int tm = 0; tm < 2; ++tm) {          // A frag: row per lane, K = 16*kh + j
            int row = wm * 32 + tm * 16 + ln;
            #pragma unroll
            for (int vg = 0; vg < 8; ++vg) {
                int kk = 16 * kh + 2 * vg;
                a[tm].u[vg] = *(const unsigned int*)&As[row * LDA + kk];
            }
        }
        #pragma unroll
        for (int tn = 0; tn < 4; ++tn) {          // B frag: col per lane, K = 16*kh + j
            int col = wn * 64 + tn * 16 + ln;
            const unsigned int* base = (const unsigned int*)&Bs[col * LDA + kh * 16];
            #pragma unroll
            for (int i = 0; i < 8; ++i) b[tn].u[i] = base[i];
        }
        #pragma unroll
        for (int tm = 0; tm < 2; ++tm)
            #pragma unroll
            for (int tn = 0; tn < 4; ++tn)
                acc[tm][tn] = __builtin_amdgcn_wmma_f32_16x16x32_bf16(
                    false, a[tm].v, false, b[tn].v, (short)0, acc[tm][tn], false, false);

        __syncthreads();
    }

    // epilogue (C layout: N = ln, M = r + 8*kh)
    #pragma unroll
    for (int tm = 0; tm < 2; ++tm) {
        #pragma unroll
        for (int tn = 0; tn < 4; ++tn) {
            int gcol = nblock + wn * 64 + tn * 16 + ln;
            #pragma unroll
            for (int r = 0; r < 8; ++r) {
                int grow = mblock + wm * 32 + tm * 16 + r + 8 * kh;
                float val = acc[tm][tn][r];
                if (mode == 0) {
                    ((float*)Cout)[(size_t)grow * N + gcol] = val;
                } else if (mode == 1) {
                    ((unsigned short*)Cout)[(size_t)grow * N + gcol] = f2bf(val);
                } else {   // per-head transposed [b][h][d][l]
                    int bb = grow / L_, l = grow % L_;
                    int hh = gcol >> 6, dd = gcol & 63;
                    ((unsigned short*)Cout)[(((size_t)bb * NH_ + hh) * D_ + dd) * L_ + l] = f2bf(val);
                }
            }
        }
    }
}

// ---------------------------------------------------------------------------
// Stage 2: flash attention. Block = 128 q rows of one (b,h); 8 waves x 16 rows.
// K tile [key][d] and V tile [d][key] (from pre-transposed Vt) staged with
// async-to-LDS copies; online softmax in C-fragment layout.
// ---------------------------------------------------------------------------
#define LDK 72   // padded LDS stride for 64-wide tiles

__global__ void __launch_bounds__(256)
attn_wmma(const unsigned short* __restrict__ Q,     // [b*L][H]   (head-major cols)
          const unsigned short* __restrict__ Kb,    // [b*L][H]
          const unsigned short* __restrict__ Vt,    // [b][h][d][key]
          const float* __restrict__ bias,           // [L][L]
          unsigned short* __restrict__ O)           // [b*L][H] bf16
{
    __shared__ __align__(16) unsigned short Ks[64 * LDK];      // [key][d]
    __shared__ __align__(16) unsigned short Vs[64 * LDK];      // [d][key]
    __shared__ __align__(16) unsigned short Ps[8][16 * LDK];   // per-wave P [q][key]

    const int t    = threadIdx.x;
    const int lane = t & 31;
    const int wave = t >> 5;
    const int kh   = lane >> 4;
    const int ln   = lane & 15;
    const int b    = blockIdx.y / NH_;
    const int h    = blockIdx.y % NH_;
    const int qbase = blockIdx.x * 128 + wave * 16;
    const float scale = 0.125f;                  // D^-0.5

    // preload Q A-fragments: D=64 contraction = 2 chunks of 32
    Frag aq[2];
    {
        const size_t qoff = (size_t)(b * L_ + qbase + ln) * H_ + h * D_;
        #pragma unroll
        for (int c = 0; c < 2; ++c)
            #pragma unroll
            for (int vg = 0; vg < 8; ++vg) {
                int dd = c * 32 + 16 * kh + 2 * vg;
                aq[c].u[vg] = *(const unsigned int*)&Q[qoff + dd];
            }
    }

    float mrow[8], lrow[8];
    v8f o[4] = {};
    #pragma unroll
    for (int r = 0; r < 8; ++r) { mrow[r] = -1e30f; lrow[r] = 0.f; }

    // cooperative-copy coordinates (each thread: 32B of K, 32B of V)
    const int crow = t >> 2;          // 0..63
    const int ccol = (t & 3) * 16;    // 0/16/32/48

    const unsigned short* kptr = &Kb[(size_t)(b * L_ + crow) * H_ + h * D_ + ccol];
    const unsigned short* vptr = &Vt[((size_t)(b * NH_ + h) * D_ + crow) * L_ + ccol];
    unsigned short* ksl = &Ks[crow * LDK + ccol];
    unsigned short* vsl = &Vs[crow * LDK + ccol];

    for (int kt = 0; kt < L_; kt += 64) {
        // --- async-stage K tile [key][d] and V tile [d][key] ---
        async_copy_b128(ksl,     kptr + (size_t)kt * H_);
        async_copy_b128(ksl + 8, kptr + (size_t)kt * H_ + 8);
        async_copy_b128(vsl,     vptr + kt);
        async_copy_b128(vsl + 8, vptr + kt + 8);
        wait_async_all();
        __syncthreads();

        // --- S = Q * K^T : 4 key tiles of 16 ---
        v8f s[4] = {};
        #pragma unroll
        for (int tn = 0; tn < 4; ++tn)
            #pragma unroll
            for (int c = 0; c < 2; ++c) {
                Frag bk;
                const unsigned int* base =
                    (const unsigned int*)&Ks[(tn * 16 + ln) * LDK + c * 32 + kh * 16];
                #pragma unroll
                for (int i = 0; i < 8; ++i) bk.u[i] = base[i];
                s[tn] = __builtin_amdgcn_wmma_f32_16x16x32_bf16(
                    false, aq[c].v, false, bk.v, (short)0, s[tn], false, false);
            }

        // --- scale + bias + online softmax (C layout: row = r+8*kh per VGPR) ---
        float pexp[4][8], mloc[8], sum8[8], alpha[8];
        #pragma unroll
        for (int r = 0; r < 8; ++r) mloc[r] = -1e30f;
        #pragma unroll
        for (int tn = 0; tn < 4; ++tn) {
            int kc = kt + tn * 16 + ln;
            #pragma unroll
            for (int r = 0; r < 8; ++r) {
                int q = qbase + r + 8 * kh;
                float val = s[tn][r] * scale + bias[(size_t)q * L_ + kc];
                pexp[tn][r] = val;
                mloc[r] = fmaxf(mloc[r], val);
            }
        }
        #pragma unroll
        for (int m = 1; m < 16; m <<= 1)     // row max over the 16-lane half
            #pragma unroll
            for (int r = 0; r < 8; ++r)
                mloc[r] = fmaxf(mloc[r], __shfl_xor(mloc[r], m, 32));
        #pragma unroll
        for (int r = 0; r < 8; ++r) {
            float mnew = fmaxf(mrow[r], mloc[r]);
            alpha[r] = __expf(mrow[r] - mnew);
            mrow[r]  = mnew;
            sum8[r]  = 0.f;
        }
        #pragma unroll
        for (int tn = 0; tn < 4; ++tn)
            #pragma unroll
            for (int r = 0; r < 8; ++r) {
                float p = __expf(pexp[tn][r] - mrow[r]);
                pexp[tn][r] = p;
                sum8[r] += p;
            }
        #pragma unroll
        for (int m = 1; m < 16; m <<= 1)     // row sum over the 16-lane half
            #pragma unroll
            for (int r = 0; r < 8; ++r)
                sum8[r] += __shfl_xor(sum8[r], m, 32);
        #pragma unroll
        for (int r = 0; r < 8; ++r) lrow[r] = lrow[r] * alpha[r] + sum8[r];
        #pragma unroll
        for (int tn = 0; tn < 4; ++tn)
            #pragma unroll
            for (int r = 0; r < 8; ++r) o[tn][r] *= alpha[r];

        // --- re-layout P (C-frag -> A-frag) via per-wave LDS slab ---
        #pragma unroll
        for (int tn = 0; tn < 4; ++tn)
            #pragma unroll
            for (int r = 0; r < 8; ++r)
                Ps[wave][(r + 8 * kh) * LDK + tn * 16 + ln] = f2bf(pexp[tn][r]);

        // --- O += P * V : contraction over 64 keys = 2 chunks of 32 ---
        #pragma unroll
        for (int c = 0; c < 2; ++c) {
            Frag ap;
            #pragma unroll
            for (int vg = 0; vg < 8; ++vg) {
                int kk = c * 32 + 16 * kh + 2 * vg;
                ap.u[vg] = *(const unsigned int*)&Ps[wave][ln * LDK + kk];
            }
            #pragma unroll
            for (int tn = 0; tn < 4; ++tn) {     // 4 d-tiles of 16
                Frag bv;
                const unsigned int* base =
                    (const unsigned int*)&Vs[(tn * 16 + ln) * LDK + c * 32 + kh * 16];
                #pragma unroll
                for (int i = 0; i < 8; ++i) bv.u[i] = base[i];
                o[tn] = __builtin_amdgcn_wmma_f32_16x16x32_bf16(
                    false, ap.v, false, bv.v, (short)0, o[tn], false, false);
            }
        }
        __syncthreads();
    }

    // normalize and write merged-head bf16 [B, L, H]
    #pragma unroll
    for (int tn = 0; tn < 4; ++tn)
        #pragma unroll
        for (int r = 0; r < 8; ++r) {
            int q  = qbase + r + 8 * kh;
            int dd = tn * 16 + ln;
            O[(size_t)(b * L_ + q) * H_ + h * D_ + dd] = f2bf(o[tn][r] / lrow[r]);
        }
}

// ---------------------------------------------------------------------------
// Launch
// ---------------------------------------------------------------------------
extern "C" void kernel_launch(void* const* d_in, const int* in_sizes, int n_in,
                              void* d_out, int out_size, void* d_ws, size_t ws_size,
                              hipStream_t stream)
{
    const float* x    = (const float*)d_in[0];
    const float* y    = (const float*)d_in[1];
    const float* bias = (const float*)d_in[2];
    const float* Wq   = (const float*)d_in[3];
    const float* Wk   = (const float*)d_in[4];
    const float* Wv   = (const float*)d_in[5];
    const float* Wo   = (const float*)d_in[6];

    unsigned short* w = (unsigned short*)d_ws;
    size_t off = 0;
    unsigned short* xh  = w + off; off += (size_t)BL_ * H_;
    unsigned short* yh  = w + off; off += (size_t)BL_ * H_;
    unsigned short* wqt = w + off; off += (size_t)H_ * H_;   // bf16, transposed [N][K]
    unsigned short* wkt = w + off; off += (size_t)H_ * H_;
    unsigned short* wvt = w + off; off += (size_t)H_ * H_;
    unsigned short* wot = w + off; off += (size_t)H_ * H_;
    unsigned short* Qb  = w + off; off += (size_t)BL_ * H_;
    unsigned short* Kb  = w + off; off += (size_t)BL_ * H_;
    unsigned short* Vt  = w + off; off += (size_t)BL_ * H_;  // [b][h][d][l]
    unsigned short* Ab  = w + off; off += (size_t)BL_ * H_;

    // Stage 0: activations -> bf16 (vectorized); weights -> bf16 transposed
    f32_to_bf16_vec<<<dim3(1024), 256, 0, stream>>>(x, xh, BL_ * H_ / 8);
    f32_to_bf16_vec<<<dim3(1024), 256, 0, stream>>>(y, yh, BL_ * H_ / 8);
    dim3 tgrid(H_ / 32, H_ / 32);   // (32, 32)
    f32_to_bf16_transpose<<<tgrid, 256, 0, stream>>>(Wq, wqt, H_, H_);
    f32_to_bf16_transpose<<<tgrid, 256, 0, stream>>>(Wk, wkt, H_, H_);
    f32_to_bf16_transpose<<<tgrid, 256, 0, stream>>>(Wv, wvt, H_, H_);
    f32_to_bf16_transpose<<<tgrid, 256, 0, stream>>>(Wo, wot, H_, H_);

    // Stage 1: projections (Q,K normal bf16; V per-head transposed)
    dim3 ggrid(H_ / 128, BL_ / 128);   // (8, 32)
    gemm_bf16_wmma<<<ggrid, 256, 0, stream>>>(xh, wqt, Qb, BL_, H_, H_, 1);
    gemm_bf16_wmma<<<ggrid, 256, 0, stream>>>(yh, wkt, Kb, BL_, H_, H_, 1);
    gemm_bf16_wmma<<<ggrid, 256, 0, stream>>>(yh, wvt, Vt, BL_, H_, H_, 2);

    // Stage 2: flash attention per (b,h), 128 q-rows per block
    attn_wmma<<<dim3(L_ / 128, B_ * NH_), 256, 0, stream>>>(Qb, Kb, Vt, bias, Ab);

    // Stage 3: output projection (f32 out)
    gemm_bf16_wmma<<<ggrid, 256, 0, stream>>>(Ab, wot, d_out, BL_, H_, H_, 0);
}
